// EdgeExtractionUniversalApproximator_28389733827000
// MI455X (gfx1250) — matrix operand order repaired and checked by reading, not compiled, not run on hardware
//
#include <hip/hip_runtime.h>

typedef _Float16 h16;
typedef __attribute__((ext_vector_type(16))) _Float16 v16h;
typedef __attribute__((ext_vector_type(8)))  _Float16 v8h;
typedef __attribute__((ext_vector_type(8)))  float    v8f;

#define NN 40000
#define NE 400000
#define MSG_LD 160   // messages row stride (floats), 145 used + zero pad

// fast sigmoid: v_exp_f32 + v_rcp_f32 (approx rcp ~1ulp; activations are f16 anyway)
__device__ __forceinline__ float sigmoidf_(float x) {
    return __builtin_amdgcn_rcpf(1.0f + __expf(-x));
}

// ---- WMMA fragment loads (CDNA5 16-bit layouts, ISA 7.12.2) -----------------
// A (16x32 f16): lane<16 -> row=lane, K = k0+0..7 and k0+16..23
//                lane>=16 -> row=lane-16, K = k0+8..15 and k0+24..31
__device__ __forceinline__ v16h load_a(const h16* X, int ld, int k0, int lane) {
    const int row = lane & 15;
    const int off = (lane & 16) ? 8 : 0;
    const h16* p = X + row * ld + k0 + off;
    v8h lo = *(const v8h*)(p);
    v8h hi = *(const v8h*)(p + 16);
    v16h a;
#pragma unroll
    for (int j = 0; j < 8; ++j) { a[j] = lo[j]; a[j + 8] = hi[j]; }
    return a;
}
// B (32x16 f16) from W stored [n][k] (row n = output col): row n0+(lane&15),
// K = k0+0..7 and k0+16..23 (lane<16), k0+8..15 and k0+24..31 (lane>=16).
__device__ __forceinline__ v16h load_b(const h16* W, int ld, int n0, int k0, int lane) {
    const int n  = n0 + (lane & 15);
    const int kb = (lane & 16) ? 8 : 0;
    const h16* p = W + n * ld + k0 + kb;
    v8h lo = *(const v8h*)(p);
    v8h hi = *(const v8h*)(p + 16);
    v16h b;
#pragma unroll
    for (int j = 0; j < 8; ++j) { b[j] = lo[j]; b[j + 8] = hi[j]; }
    return b;
}

template <int KP>
__device__ __forceinline__ void load_a_all(const h16* X, int ldx, int lane, v16h* a) {
#pragma unroll
    for (int kk = 0; kk < KP / 32; ++kk) a[kk] = load_a(X, ldx, kk * 32, lane);
}

// two independent accumulator chains (N-tiles n0 and n0+16) so B loads for one
// WMMA overlap execution of the other
template <int KP>
__device__ __forceinline__ void gemm_pair(const v16h* a, const h16* W, int n0, int lane,
                                          v8f& c0, v8f& c1) {
#pragma unroll
    for (int kk = 0; kk < KP / 32; ++kk) {
        v16h b0 = load_b(W, KP, n0,      kk * 32, lane);
        v16h b1 = load_b(W, KP, n0 + 16, kk * 32, lane);
        c0 = __builtin_amdgcn_wmma_f32_16x16x32_f16(false, a[kk], false, b0, (short)0, c0, false, false);
        c1 = __builtin_amdgcn_wmma_f32_16x16x32_f16(false, a[kk], false, b1, (short)0, c1, false, false);
    }
}

// ---- block-cooperative weight staging: fp32 global -> zero-padded f16 LDS ---
template <int Npad, int Kpad>
__device__ __forceinline__ void stage_w(h16* WL, float* BL, const float* gw, const float* gb,
                                        int Nout, int Kin) {
    __syncthreads();
    const int t = threadIdx.x, nt = blockDim.x;
    for (int i = t; i < Npad * Kpad; i += nt) {
        int n = i / Kpad, k = i - n * Kpad;
        WL[i] = (h16)((n < Nout && k < Kin) ? gw[n * Kin + k] : 0.0f);
    }
    for (int i = t; i < Npad; i += nt) BL[i] = (i < Nout) ? gb[i] : 0.0f;
    __syncthreads();
}

// ---- one dense layer for a 16-row tile:  OUT = act(X @ W^T + b) -------------
// ACT: 0 linear, 1 relu, 2 sigmoid.  ACCUM: OUT += act(...) instead of store.
template <int ACT, bool ACCUM, int KP, int NP>
__device__ __forceinline__ void wave_layer(const h16* X, int ldx, const h16* W, const float* B,
                                           h16* OUT, int ldo, int lane) {
    const int colL = lane & 15;
    const int rb   = (lane & 16) ? 8 : 0;
    v16h a[KP / 32];
    load_a_all<KP>(X, ldx, lane, a);
#pragma unroll
    for (int n0 = 0; n0 < NP; n0 += 32) {
        v8f c0, c1;
        {
            float bv0 = B[n0 + colL], bv1 = B[n0 + 16 + colL];
#pragma unroll
            for (int i = 0; i < 8; ++i) { c0[i] = bv0; c1[i] = bv1; }
        }
        gemm_pair<KP>(a, W, n0, lane, c0, c1);
        const int col0 = n0 + colL, col1 = n0 + 16 + colL;
#pragma unroll
        for (int i = 0; i < 8; ++i) {
            float v0 = c0[i], v1 = c1[i];
            if (ACT == 1) { v0 = fmaxf(v0, 0.0f); v1 = fmaxf(v1, 0.0f); }
            if (ACT == 2) { v0 = sigmoidf_(v0);   v1 = sigmoidf_(v1); }
            h16* q0 = OUT + (rb + i) * ldo + col0;
            h16* q1 = OUT + (rb + i) * ldo + col1;
            if (ACCUM) { *q0 = (h16)((float)(*q0) + v0); *q1 = (h16)((float)(*q1) + v1); }
            else       { *q0 = (h16)v0; *q1 = (h16)v1; }
        }
    }
}

// ---- kernel 0: zero the message accumulator ---------------------------------
__global__ void k_zero(float* p, int n) {
    for (int i = blockIdx.x * blockDim.x + threadIdx.x; i < n; i += gridDim.x * blockDim.x)
        p[i] = 0.0f;
}

// ---- kernel 1: sub = sigmoid(mlp1([nf[src[src]], er, ea])); segment_sum -----
__global__ __launch_bounds__(256) void k_msg(const float* __restrict__ nf,
                                             const float* __restrict__ er,
                                             const float* __restrict__ ea,
                                             const int* __restrict__ ei,
                                             const float* __restrict__ w, const float* __restrict__ b,
                                             float* __restrict__ messages) {
    extern __shared__ char smem[];
    h16*   WL = (h16*)smem;                       // 160*160 halves
    float* BL = (float*)(WL + 160 * 160);         // 160 floats
    h16*   XB = (h16*)(BL + 160);                 // 8 waves * 16*160 halves
    const int lane = threadIdx.x & 31;
    const int wid  = threadIdx.x >> 5;
    h16* X = XB + wid * 16 * 160;
    const int* src = ei;
    const int* dst = ei + NE;
    const int tile = blockIdx.x * 8 + wid;
    const int e0   = tile * 16;

    const int e_r = e0 + (lane & 15);
    const int s1  = src[e_r];
    const int g   = src[s1];       // reference quirk: node_features[src[src]]
    const int d   = dst[e_r];

    for (int i = lane; i < 16 * 160; i += 32) {
        int r = i / 160, c = i - r * 160;
        float v;
        if (c < 128)      { int gr = __shfl(g, r, 32); v = nf[gr * 128 + c]; }
        else if (c < 136)   v = er[(e0 + r) * 8 + (c - 128)];
        else if (c < 145)   v = ea[(e0 + r) * 9 + (c - 136)];
        else                v = 0.0f;
        X[i] = (h16)v;
    }

    stage_w<160, 160>(WL, BL, w, b, 145, 145);

    const int colL = lane & 15;
    const int rb   = (lane & 16) ? 8 : 0;
    v16h a[5];
    load_a_all<160>(X, 160, lane, a);
#pragma unroll
    for (int n0 = 0; n0 < 160; n0 += 32) {
        v8f c0, c1;
        {
            float bv0 = BL[n0 + colL], bv1 = BL[n0 + 16 + colL];
#pragma unroll
            for (int i = 0; i < 8; ++i) { c0[i] = bv0; c1[i] = bv1; }
        }
        gemm_pair<160>(a, WL, n0, lane, c0, c1);
        const int col0 = n0 + colL, col1 = n0 + 16 + colL;   // col0 <= 143 always
#pragma unroll
        for (int i = 0; i < 8; ++i) {
            int dr = __shfl(d, rb + i, 32);
            atomicAdd(&messages[dr * MSG_LD + col0], sigmoidf_(c0[i]));   // fp32 segment_sum
            if (col1 < 145)
                atomicAdd(&messages[dr * MSG_LD + col1], sigmoidf_(c1[i]));
        }
    }
}

// ---- kernel 2: messages->mlp2; node_upd = sigmoid(self(nf)+neigh(m)) + nf ---
__global__ __launch_bounds__(256) void k_node(const float* __restrict__ nf,
                                              const float* __restrict__ messages,
                                              const float* w0, const float* b0,
                                              const float* w1, const float* b1,
                                              const float* w2, const float* b2,
                                              const float* w3, const float* b3,
                                              const float* sw, const float* sb,
                                              const float* nw, const float* nb_,
                                              h16* __restrict__ node_upd) {
    extern __shared__ char smem[];
    h16*   WL = (h16*)smem;                       // 32768 halves (self+neigh fit)
    float* BL = (float*)(WL + 32768);             // 2*160 floats
    h16*   XB = (h16*)(BL + 320);
    const int lane = threadIdx.x & 31;
    const int wid  = threadIdx.x >> 5;
    h16* A  = XB + wid * 2 * 16 * 160;
    h16* Bb = A + 16 * 160;
    const int tile = blockIdx.x * 8 + wid;
    const int nbse = tile * 16;

    for (int i = lane; i < 16 * 160; i += 32) {
        int r = i / 160, c = i - r * 160;
        int node = nbse + r; if (node >= NN) node = NN - 1;
        A[i] = (h16)messages[node * MSG_LD + c];
    }
    stage_w<96, 160>(WL, BL, w0, b0, 88, 145);  wave_layer<1, false, 160, 96>(A, 160, WL, BL, Bb, 160, lane);
    stage_w<32, 96>(WL, BL, w1, b1, 32, 88);    wave_layer<1, false, 96, 32>(Bb, 160, WL, BL, A, 160, lane);
    stage_w<64, 32>(WL, BL, w2, b2, 64, 32);    wave_layer<1, false, 32, 64>(A, 160, WL, BL, Bb, 160, lane);
    stage_w<128, 64>(WL, BL, w3, b3, 128, 64);  wave_layer<0, false, 64, 128>(Bb, 160, WL, BL, A, 160, lane);
    // A = m (16x128).  Bb <- nf rows (f16 copy, used only as GEMM input)
    __syncthreads();
    for (int i = lane; i < 16 * 128; i += 32) {
        int r = i / 128, c = i - r * 128;
        int node = nbse + r; if (node >= NN) node = NN - 1;
        Bb[r * 160 + c] = (h16)nf[node * 128 + c];
    }
    __syncthreads();
    {   // stage self_w and neigh_w together
        const int t = threadIdx.x, nt = blockDim.x;
        for (int i = t; i < 128 * 128; i += nt) { WL[i] = (h16)sw[i]; WL[16384 + i] = (h16)nw[i]; }
        for (int i = t; i < 128; i += nt)       { BL[i] = sb[i]; BL[160 + i] = nb_[i]; }
    }
    __syncthreads();
    const int colL = lane & 15;
    const int rb   = (lane & 16) ? 8 : 0;
    v16h af[4], am[4];
    load_a_all<128>(Bb, 160, lane, af);   // nf
    load_a_all<128>(A, 160, lane, am);    // m
#pragma unroll
    for (int n0 = 0; n0 < 128; n0 += 32) {
        v8f c0, c1;
        {
            float bv0 = BL[n0 + colL] + BL[160 + n0 + colL];
            float bv1 = BL[n0 + 16 + colL] + BL[160 + n0 + 16 + colL];
#pragma unroll
            for (int i = 0; i < 8; ++i) { c0[i] = bv0; c1[i] = bv1; }
        }
        gemm_pair<128>(af, WL, n0, lane, c0, c1);           // self(nf)
        gemm_pair<128>(am, WL + 16384, n0, lane, c0, c1);   // neigh(m)
        const int col0 = n0 + colL, col1 = n0 + 16 + colL;
#pragma unroll
        for (int i = 0; i < 8; ++i) {
            int node = nbse + rb + i;
            if (node < NN) {
                float x0 = nf[node * 128 + col0];    // exact fp32 residual add
                float x1 = nf[node * 128 + col1];
                node_upd[node * 128 + col0] = (h16)(sigmoidf_(c0[i]) + x0);
                node_upd[node * 128 + col1] = (h16)(sigmoidf_(c1[i]) + x1);
            }
        }
    }
}

// ---- helper: build [node_upd[idx], er, ea] tile (cols 0..159) ---------------
__device__ __forceinline__ void fill_edge_in(h16* X, int idx, const h16* nu,
                                             const float* er, const float* ea,
                                             int e0, int lane) {
    __syncthreads();
    for (int i = lane; i < 16 * 160; i += 32) {
        int r = i / 160, c = i - r * 160;
        h16 v;
        if (c < 128)      { int nr = __shfl(idx, r, 32); v = nu[nr * 128 + c]; }
        else if (c < 136)   v = (h16)er[(e0 + r) * 8 + (c - 128)];
        else if (c < 145)   v = (h16)ea[(e0 + r) * 9 + (c - 136)];
        else                v = (h16)0.0f;
        X[r * 288 + c] = v;
    }
}

// ---- kernel 3: edge update MLPs + head MLP, fully fused per edge tile -------
__global__ __launch_bounds__(256) void k_edge(const h16* __restrict__ nu,
                                              const float* __restrict__ er,
                                              const float* __restrict__ ea,
                                              const int* __restrict__ ei,
                                              const float* e1w, const float* e1b,
                                              const float* m1w, const float* m1b,
                                              const float* e2w0, const float* e2b0,
                                              const float* e2w1, const float* e2b1,
                                              const float* e2w2, const float* e2b2,
                                              const float* e2w3, const float* e2b3,
                                              const float* hw0, const float* hb0,
                                              const float* hw1, const float* hb1,
                                              const float* hw2, const float* hb2,
                                              const float* hw3, const float* hb3,
                                              const float* hw4, const float* hb4,
                                              float* __restrict__ out) {
    extern __shared__ char smem[];
    h16*   WL = (h16*)smem;                       // 128*288 = 36864 halves max
    float* BL = (float*)(WL + 36864);             // 160 floats
    h16*   XB = (h16*)(BL + 160);
    const int lane = threadIdx.x & 31;
    const int wid  = threadIdx.x >> 5;
    h16* A  = XB + wid * 2 * 16 * 288;
    h16* Bb = A + 16 * 288;
    const int* src = ei;
    const int* dst = ei + NE;
    const int tile = blockIdx.x * 8 + wid;
    const int e0   = tile * 16;
    const int s = src[e0 + (lane & 15)];
    const int d = dst[e0 + (lane & 15)];

    // t = sigmoid(emp1(src_in)) + sigmoid(mp1(dst_in))   -> Bb
    fill_edge_in(A, s, nu, er, ea, e0, lane);
    stage_w<160, 160>(WL, BL, e1w, e1b, 145, 145);
    wave_layer<2, false, 160, 160>(A, 288, WL, BL, Bb, 288, lane);
    fill_edge_in(A, d, nu, er, ea, e0, lane);
    stage_w<160, 160>(WL, BL, m1w, m1b, 145, 145);
    wave_layer<2, true, 160, 160>(A, 288, WL, BL, Bb, 288, lane);
    // emlp2 chain: 145->74->4->8->17
    stage_w<96, 160>(WL, BL, e2w0, e2b0, 74, 145);  wave_layer<1, false, 160, 96>(Bb, 288, WL, BL, A, 288, lane);
    stage_w<32, 96>(WL, BL, e2w1, e2b1, 4, 74);     wave_layer<1, false, 96, 32>(A, 288, WL, BL, Bb, 288, lane);
    stage_w<32, 32>(WL, BL, e2w2, e2b2, 8, 4);      wave_layer<1, false, 32, 32>(Bb, 288, WL, BL, A, 288, lane);
    stage_w<32, 32>(WL, BL, e2w3, e2b3, 17, 8);     wave_layer<0, false, 32, 32>(A, 288, WL, BL, Bb, 288, lane);
    // head_in = [nu[src], nu[dst], edge_upd] -> A (16x288, cols 273..287 zero)
    __syncthreads();
    for (int i = lane; i < 16 * 288; i += 32) {
        int r = i / 288, c = i - r * 288;
        h16 v;
        if (c < 128)      { int nr = __shfl(s, r, 32); v = nu[nr * 128 + c]; }
        else if (c < 256) { int nr = __shfl(d, r, 32); v = nu[nr * 128 + (c - 128)]; }
        else if (c < 273)   v = Bb[r * 288 + (c - 256)];
        else                v = (h16)0.0f;
        A[i] = v;
    }
    // head: 273->128->128->128->128->81
    stage_w<128, 288>(WL, BL, hw0, hb0, 128, 273);  wave_layer<1, false, 288, 128>(A, 288, WL, BL, Bb, 288, lane);
    stage_w<128, 128>(WL, BL, hw1, hb1, 128, 128);  wave_layer<1, false, 128, 128>(Bb, 288, WL, BL, A, 288, lane);
    stage_w<128, 128>(WL, BL, hw2, hb2, 128, 128);  wave_layer<1, false, 128, 128>(A, 288, WL, BL, Bb, 288, lane);
    stage_w<128, 128>(WL, BL, hw3, hb3, 128, 128);  wave_layer<1, false, 128, 128>(Bb, 288, WL, BL, A, 288, lane);
    stage_w<96, 128>(WL, BL, hw4, hb4, 81, 128);
    const int colL = lane & 15;
    const int rb   = (lane & 16) ? 8 : 0;
    v16h a[4];
    load_a_all<128>(A, 288, lane, a);
#pragma unroll
    for (int n0 = 0; n0 < 96; n0 += 32) {
        v8f c0, c1;
        {
            float bv0 = BL[n0 + colL], bv1 = BL[n0 + 16 + colL];
#pragma unroll
            for (int i = 0; i < 8; ++i) { c0[i] = bv0; c1[i] = bv1; }
        }
        gemm_pair<128>(a, WL, n0, lane, c0, c1);
        const int col0 = n0 + colL, col1 = n0 + 16 + colL;   // col0 <= 79 always
#pragma unroll
        for (int i = 0; i < 8; ++i) {
            out[(size_t)(e0 + rb + i) * 81 + col0] = c0[i];
            if (col1 < 81)
                out[(size_t)(e0 + rb + i) * 81 + col1] = c1[i];
        }
    }
}

extern "C" void kernel_launch(void* const* d_in, const int* in_sizes, int n_in,
                              void* d_out, int out_size, void* d_ws, size_t ws_size,
                              hipStream_t stream) {
    (void)in_sizes; (void)n_in; (void)out_size; (void)ws_size;
    const float* nf = (const float*)d_in[0];
    const float* er = (const float*)d_in[1];
    const float* ea = (const float*)d_in[2];
    const int*   ei = (const int*)d_in[3];
    const float *mp1w = (const float*)d_in[5],  *mp1b = (const float*)d_in[6];
    const float *w0 = (const float*)d_in[7],  *b0 = (const float*)d_in[8];
    const float *w1 = (const float*)d_in[9],  *b1 = (const float*)d_in[10];
    const float *w2 = (const float*)d_in[11], *b2 = (const float*)d_in[12];
    const float *w3 = (const float*)d_in[13], *b3 = (const float*)d_in[14];
    const float *sw = (const float*)d_in[15], *sb = (const float*)d_in[16];
    const float *nw = (const float*)d_in[17], *nb = (const float*)d_in[18];
    const float *e1w = (const float*)d_in[19], *e1b = (const float*)d_in[20];
    const float *e2w0 = (const float*)d_in[21], *e2b0 = (const float*)d_in[22];
    const float *e2w1 = (const float*)d_in[23], *e2b1 = (const float*)d_in[24];
    const float *e2w2 = (const float*)d_in[25], *e2b2 = (const float*)d_in[26];
    const float *e2w3 = (const float*)d_in[27], *e2b3 = (const float*)d_in[28];
    const float *hw0 = (const float*)d_in[29], *hb0 = (const float*)d_in[30];
    const float *hw1 = (const float*)d_in[31], *hb1 = (const float*)d_in[32];
    const float *hw2 = (const float*)d_in[33], *hb2 = (const float*)d_in[34];
    const float *hw3 = (const float*)d_in[35], *hb3 = (const float*)d_in[36];
    const float *hw4 = (const float*)d_in[37], *hb4 = (const float*)d_in[38];

    float* messages = (float*)d_ws;                                    // 40000*160 f32
    h16*   node_upd = (h16*)((char*)d_ws + (size_t)NN * MSG_LD * 4);   // 40000*128 f16
    float* out = (float*)d_out;

    k_zero<<<1024, 256, 0, stream>>>(messages, NN * MSG_LD);

    size_t sm1 = 160 * 160 * 2 + 160 * 4 + 8 * 16 * 160 * 2;               // ~93 KB
    k_msg<<<NE / 128, 256, sm1, stream>>>(nf, er, ea, ei, mp1w, mp1b, messages);

    size_t smn = 32768 * 2 + 320 * 4 + 8 * 2 * 16 * 160 * 2;               // ~149 KB
    k_node<<<(NN + 127) / 128, 256, smn, stream>>>(nf, messages,
                                                   w0, b0, w1, b1, w2, b2, w3, b3,
                                                   sw, sb, nw, nb, node_upd);

    size_t sm2 = 36864 * 2 + 160 * 4 + 8 * 2 * 16 * 288 * 2;               // ~222 KB
    k_edge<<<NE / 128, 256, sm2, stream>>>(node_upd, er, ea, ei,
                                           e1w, e1b, mp1w, mp1b,
                                           e2w0, e2b0, e2w1, e2b1, e2w2, e2b2, e2w3, e2b3,
                                           hw0, hb0, hw1, hb1, hw2, hb2, hw3, hb3, hw4, hb4,
                                           out);
}
